// SCAgentDecoder_74775380623816
// MI455X (gfx1250) — matrix operand order repaired and checked by reading, not compiled, not run on hardware
//
#include <hip/hip_runtime.h>

#define H 256
#define NHD 8
#define HDIM 32
#define FF 64
#define NA 512
#define NSTEP 32
#define NTOK (NA*NSTEP)
#define NTOKEN 1024
#define FLDA 136   // fourier feature row stride (bf16 elems, 16B-aligned rows)
#define PI_F 3.14159265358979323846f
#define INV_SQRT_HD 0.17677669529663688f

typedef __bf16 bf16;
typedef __attribute__((ext_vector_type(16))) __bf16 bfx16;
typedef __attribute__((ext_vector_type(8)))  __bf16 bfx8;
typedef __attribute__((ext_vector_type(8)))  float  fx8;

// ------------------------------------------------------------------
// Device helpers
// ------------------------------------------------------------------
__device__ __forceinline__ float wrap_pi(float a) {
  float w = fmodf(a + PI_F, 2.0f * PI_F);
  w = (w < 0.0f) ? w + 2.0f * PI_F : w;
  return w - PI_F;
}
__device__ __forceinline__ float angle2d(float cx, float cy, float nx, float ny) {
  float cr = cx * ny - cy * nx;
  float dt = cx * nx + cy * ny;
  bool safe = (cr * cr + dt * dt) > 1e-12f;
  return safe ? atan2f(cr, dt) : 0.0f;
}

// gfx1250 async global->LDS 16B copy (ASYNCcnt-tracked, no VGPR round trip).
// lds_off: allocation-relative LDS byte offset; gp: 64-bit global address.
__device__ __forceinline__ void scad_async_b128(unsigned lds_off, const void* gp) {
  asm volatile("global_load_async_to_lds_b128 %0, %1, off"
               :: "v"(lds_off), "v"(gp) : "memory");
}
__device__ __forceinline__ void scad_wait_async() {
  asm volatile("s_wait_asynccnt 0" ::: "memory");
}

// ------------------------------------------------------------------
// Tiled GEMM: C[M,N] (+)= A[M,K] @ B[K,N] (+bias) (relu)
// A: f32 or bf16 (ABF, compile-time). B: f32 weights. C: f32 or bf16 (CBF).
// Tile: 128x64, 8 waves, each wave 32x32 = 2x2 WMMA(16x16x32 bf16, f32 acc).
// Full interior tiles take a branchless b128 path (bf16 A goes through the
// CDNA5 async global->LDS copy); edges take a scalar bounds-checked path.
// flags: 1=bias, 2=relu, 4=accumulate-into-C
// ------------------------------------------------------------------
#define KT 32
#define LDR 40   // KT + 8 bf16 padding (80B rows, 16B aligned)

template<int ABF, int CBF>
__global__ __launch_bounds__(256) void scad_gemm_kernel(
    const void* __restrict__ Ap, long lda,
    const float* __restrict__ B, long ldb,
    const float* __restrict__ bias,
    void* __restrict__ Cp, long ldc,
    int M, int N, int K, int flags)
{
  __shared__ __align__(16) bf16 As[128 * LDR];
  __shared__ __align__(16) bf16 Bs[64 * LDR];
  const int t = threadIdx.x;
  const int bm = blockIdx.y * 128, bn = blockIdx.x * 64;
  const int w = t >> 5, lane = t & 31;
  const int wm = w >> 1, wn = w & 1;
  const int lm = lane & 15, lh = lane >> 4;
  const bool mnfull = (bm + 128 <= M) && (bn + 64 <= N);
  fx8 acc[2][2] = {};
  const float* Af = (const float*)Ap;
  const bf16*  Ab = (const bf16*)Ap;

  for (int kb = 0; kb < K; kb += KT) {
    __syncthreads();
    if (mnfull && (kb + KT <= K)) {
      // ---- fast path: branchless b128 staging ----
      if (ABF) {
        // pure copy -> CDNA5 async global->LDS (B128), ASYNCcnt tracked
        #pragma unroll
        for (int i = 0; i < 2; ++i) {
          int e = (i << 8) + t;                 // 512 chunks of 8 bf16
          int r = e >> 2, c8 = (e & 3) << 3;
          unsigned lo = (unsigned)(uintptr_t)&As[r * LDR + c8];
          scad_async_b128(lo, Ab + (long)(bm + r) * lda + kb + c8);
        }
      } else {
        float4 va[4];
        #pragma unroll
        for (int i = 0; i < 4; ++i) {
          int e = (i << 8) + t;                 // 1024 float4 chunks
          int r = e >> 3, c4 = (e & 7) << 2;
          va[i] = *(const float4*)(Af + (long)(bm + r) * lda + kb + c4);
        }
        #pragma unroll
        for (int i = 0; i < 4; ++i) {
          int e = (i << 8) + t;
          int r = e >> 3, c4 = (e & 7) << 2;
          union { bf16 h[4]; uint2 u; } pk;
          pk.h[0] = (bf16)va[i].x; pk.h[1] = (bf16)va[i].y;
          pk.h[2] = (bf16)va[i].z; pk.h[3] = (bf16)va[i].w;
          *(uint2*)&As[r * LDR + c4] = pk.u;
        }
      }
      {
        float4 vB[2];
        #pragma unroll
        for (int i = 0; i < 2; ++i) {
          int e = (i << 8) + t;                 // 512 float4 chunks along n
          int kk = e >> 4, n4 = (e & 15) << 2;
          vB[i] = *(const float4*)(B + (long)(kb + kk) * ldb + bn + n4);
        }
        #pragma unroll
        for (int i = 0; i < 2; ++i) {
          int e = (i << 8) + t;
          int kk = e >> 4, n4 = (e & 15) << 2;
          Bs[(n4 + 0) * LDR + kk] = (bf16)vB[i].x;
          Bs[(n4 + 1) * LDR + kk] = (bf16)vB[i].y;
          Bs[(n4 + 2) * LDR + kk] = (bf16)vB[i].z;
          Bs[(n4 + 3) * LDR + kk] = (bf16)vB[i].w;
        }
      }
      if (ABF) scad_wait_async();
    } else {
      // ---- edge path: scalar, bounds checked ----
      #pragma unroll
      for (int i = 0; i < 16; ++i) {
        int e = (i << 8) + t;
        int r = e >> 5, c = e & 31;
        int gr = bm + r, gk = kb + c;
        float v = 0.0f;
        if (gr < M && gk < K) {
          if (ABF) v = (float)Ab[(long)gr * lda + gk];
          else     v = Af[(long)gr * lda + gk];
        }
        As[r * LDR + c] = (bf16)v;
      }
      #pragma unroll
      for (int i = 0; i < 8; ++i) {
        int e = (i << 8) + t;
        int n = e & 63, kk = e >> 6;
        int gn = bn + n, gk = kb + kk;
        float v = 0.0f;
        if (gn < N && gk < K) v = B[(long)gk * ldb + gn];
        Bs[n * LDR + kk] = (bf16)v;
      }
    }
    __syncthreads();

    union FRU { bfx16 v; bfx8 h[2]; };
    FRU afr[2], bfr[2];
    // A frag (16x32): lanes 0-15 row m, K {0..7,16..23}; lanes 16-31 K {8..15,24..31}
    #pragma unroll
    for (int mi = 0; mi < 2; ++mi) {
      const bf16* p = &As[(wm * 32 + mi * 16 + lm) * LDR + lh * 8];
      afr[mi].h[0] = *(const bfx8*)(p);
      afr[mi].h[1] = *(const bfx8*)(p + 16);
    }
    // B frag (32x16): lanes 0-15 col n, K {0..7,16..23}; lanes 16-31 K {8..15,24..31}
    #pragma unroll
    for (int ni = 0; ni < 2; ++ni) {
      const bf16* p = &Bs[(wn * 32 + ni * 16 + lm) * LDR + lh * 8];
      bfr[ni].h[0] = *(const bfx8*)(p);
      bfr[ni].h[1] = *(const bfx8*)(p + 16);
    }
    #pragma unroll
    for (int mi = 0; mi < 2; ++mi)
      #pragma unroll
      for (int ni = 0; ni < 2; ++ni)
        acc[mi][ni] = __builtin_amdgcn_wmma_f32_16x16x32_bf16(
            false, afr[mi].v, false, bfr[ni].v, (short)0, acc[mi][ni], false, false);
  }

  // Epilogue. C/D layout: col = lane&15, row_in_tile = r + 8*(lane>>4)
  float* Cf = (float*)Cp;
  bf16*  Cb = (bf16*)Cp;
  if (mnfull) {
    #pragma unroll
    for (int ni = 0; ni < 2; ++ni) {
      int col = bn + wn * 32 + ni * 16 + lm;
      float bv = (flags & 1) ? bias[col] : 0.0f;
      #pragma unroll
      for (int mi = 0; mi < 2; ++mi) {
        #pragma unroll
        for (int r = 0; r < 8; ++r) {
          int row = bm + wm * 32 + mi * 16 + lh * 8 + r;
          long ci = (long)row * ldc + col;
          float v = acc[mi][ni][r] + bv;
          if (flags & 4) v += CBF ? (float)Cb[ci] : Cf[ci];
          if (flags & 2) v = fmaxf(v, 0.0f);
          if (CBF) Cb[ci] = (bf16)v;
          else     Cf[ci] = v;
        }
      }
    }
  } else {
    #pragma unroll
    for (int ni = 0; ni < 2; ++ni) {
      int col = bn + wn * 32 + ni * 16 + lm;
      if (col >= N) continue;
      float bv = (flags & 1) ? bias[col] : 0.0f;
      #pragma unroll
      for (int mi = 0; mi < 2; ++mi) {
        #pragma unroll
        for (int r = 0; r < 8; ++r) {
          int row = bm + wm * 32 + mi * 16 + lh * 8 + r;
          if (row >= M) continue;
          long ci = (long)row * ldc + col;
          float v = acc[mi][ni][r] + bv;
          if (flags & 4) v += CBF ? (float)Cb[ci] : Cf[ci];
          if (flags & 2) v = fmaxf(v, 0.0f);
          if (CBF) Cb[ci] = (bf16)v;
          else     Cf[ci] = v;
        }
      }
    }
  }
}

// ------------------------------------------------------------------
// LayerNorm over H=256, one wave per row, 8 rows/block. relu optional.
// ------------------------------------------------------------------
template<int IBF, int OBF>
__global__ __launch_bounds__(256) void scad_ln_kernel(
    const void* __restrict__ inp, void* __restrict__ outp,
    const float* __restrict__ g, const float* __restrict__ b,
    int M, int relu)
{
  int w = threadIdx.x >> 5, lane = threadIdx.x & 31;
  long row = (long)blockIdx.x * 8 + w;
  if (row >= M) return;
  const float* inf = (const float*)inp + row * H;
  const bf16*  inb = (const bf16*)inp + row * H;
  float x[8];
  float s = 0.0f;
  #pragma unroll
  for (int j = 0; j < 8; ++j) {
    int c = j * 32 + lane;
    x[j] = IBF ? (float)inb[c] : inf[c];
    s += x[j];
  }
  #pragma unroll
  for (int o = 16; o > 0; o >>= 1) s += __shfl_xor(s, o);
  float mean = s * (1.0f / H);
  float vs = 0.0f;
  #pragma unroll
  for (int j = 0; j < 8; ++j) { float d = x[j] - mean; vs += d * d; }
  #pragma unroll
  for (int o = 16; o > 0; o >>= 1) vs += __shfl_xor(vs, o);
  float rstd = rsqrtf(vs * (1.0f / H) + 1e-5f);
  float* outf = (float*)outp + row * H;
  bf16*  outb = (bf16*)outp + row * H;
  #pragma unroll
  for (int j = 0; j < 8; ++j) {
    int c = j * 32 + lane;
    float y = (x[j] - mean) * rstd * g[c] + b[c];
    if (relu) y = fmaxf(y, 0.0f);
    if (OBF) outb[c] = (bf16)y;
    else     outf[c] = y;
  }
}

// ------------------------------------------------------------------
// Elementwise / feature kernels
// ------------------------------------------------------------------
__global__ __launch_bounds__(256) void scad_feat_kernel(
    const float* __restrict__ pos, const float* __restrict__ head, float* __restrict__ feat)
{
  int i = blockIdx.x * 256 + threadIdx.x;
  if (i >= NTOK) return;
  int s = i & (NSTEP - 1);
  float mx = 0.0f, my = 0.0f;
  if (s > 0) { mx = pos[2 * i] - pos[2 * i - 2]; my = pos[2 * i + 1] - pos[2 * i - 1]; }
  float h = head[i];
  float hx = cosf(h), hy = sinf(h);
  feat[2 * i]     = sqrtf(mx * mx + my * my + 1e-10f);
  feat[2 * i + 1] = angle2d(hx, hy, mx, my);
}

__global__ __launch_bounds__(256) void scad_gather_tok_kernel(
    const float* __restrict__ tok_all, const int* __restrict__ ati, float* __restrict__ out)
{
  long i = (long)blockIdx.x * 256 + threadIdx.x;
  if (i >= (long)NTOK * H) return;
  long row = i >> 8; int c = (int)(i & 255);
  out[i] = tok_all[(long)ati[row] * H + c];
}

__global__ __launch_bounds__(256) void scad_fourier_feat_kernel(
    const float* __restrict__ X, int ldx, int d, const float* __restrict__ freqs,
    bf16* __restrict__ out, int M)
{
  long i = (long)blockIdx.x * 256 + threadIdx.x;
  if (i >= (long)M * FF) return;
  long row = i >> 6; int f = (int)(i & 63);
  float x = X[row * ldx + d];
  float ang = x * freqs[f] * (2.0f * PI_F);
  bf16* o = out + row * FLDA;
  o[f]      = (bf16)cosf(ang);
  o[64 + f] = (bf16)sinf(ang);
  if (f == 0) o[128] = (bf16)x;
}

__global__ __launch_bounds__(256) void scad_add_cat_kernel(
    float* __restrict__ s, const float* __restrict__ type_emb, const float* __restrict__ state_emb,
    const float* __restrict__ owner_emb, const float* __restrict__ props_out,
    const int* __restrict__ atype, const int* __restrict__ ustate, const int* __restrict__ owner)
{
  long i = (long)blockIdx.x * 256 + threadIdx.x;
  if (i >= (long)NTOK * H) return;
  long row = i >> 8; int c = (int)(i & 255);
  int a = (int)(row >> 5);
  s[i] += type_emb[(long)atype[a] * H + c] + state_emb[(long)ustate[a] * H + c]
        + owner_emb[(long)owner[a] * H + c] + props_out[(long)a * H + c];
}

__global__ __launch_bounds__(256) void scad_edge_feat_t_kernel(
    const float* __restrict__ pos, const float* __restrict__ head,
    const int* __restrict__ src, const int* __restrict__ dst, int E, float* __restrict__ rt)
{
  int e = blockIdx.x * 256 + threadIdx.x;
  if (e >= E) return;
  int s = src[e], d = dst[e];
  float rx = pos[2 * s] - pos[2 * d], ry = pos[2 * s + 1] - pos[2 * d + 1];
  float hd = head[d];
  float hx = cosf(hd), hy = sinf(hd);
  rt[4 * e]     = sqrtf(rx * rx + ry * ry + 1e-10f);
  rt[4 * e + 1] = angle2d(hx, hy, rx, ry);
  rt[4 * e + 2] = wrap_pi(head[s] - hd);
  rt[4 * e + 3] = (float)(s - d);
}

__global__ __launch_bounds__(256) void scad_edge_feat_a2a_kernel(
    const float* __restrict__ pos, const float* __restrict__ head,
    const int* __restrict__ src, const int* __restrict__ dst, int E,
    float* __restrict__ ra, int* __restrict__ srcA, int* __restrict__ dstA)
{
  int e = blockIdx.x * 256 + threadIdx.x;
  if (e >= E) return;
  int s = src[e], d = dst[e];
  // step-major id -> agent-major id
  int si = (s % NA) * NSTEP + (s / NA);
  int di = (d % NA) * NSTEP + (d / NA);
  srcA[e] = si; dstA[e] = di;
  float rx = pos[2 * si] - pos[2 * di], ry = pos[2 * si + 1] - pos[2 * di + 1];
  float hd = head[di];
  float hx = cosf(hd), hy = sinf(hd);
  ra[3 * e]     = sqrtf(rx * rx + ry * ry + 1e-10f);
  ra[3 * e + 1] = angle2d(hx, hy, rx, ry);
  ra[3 * e + 2] = wrap_pi(head[si] - hd);
}

__global__ __launch_bounds__(256) void scad_zero_i32_kernel(int* __restrict__ p, long n)
{
  long i = (long)blockIdx.x * 256 + threadIdx.x;
  if (i < n) p[i] = 0;
}

__global__ __launch_bounds__(256) void scad_csr_build_kernel(
    const int* __restrict__ dst, int E, int* __restrict__ cnt, int* __restrict__ idx)
{
  int e = blockIdx.x * 256 + threadIdx.x;
  if (e >= E) return;
  int d = dst[e];
  int s = atomicAdd(&cnt[d], 1);
  if (s < 8) idx[(long)d * 8 + s] = e;
}

__global__ __launch_bounds__(256) void scad_sim_kernel(
    const float* __restrict__ q, const float* __restrict__ kn, const bf16* __restrict__ kr,
    const int* __restrict__ src, const int* __restrict__ dst, int E, float* __restrict__ sim)
{
  long id = (long)blockIdx.x * 256 + threadIdx.x;
  if (id >= (long)E * NHD) return;
  long e = id >> 3; int h = (int)(id & 7);
  const float* qp = q  + (long)dst[e] * H + h * HDIM;
  const float* kp = kn + (long)src[e] * H + h * HDIM;
  const bf16*  rp = kr + e * H + h * HDIM;
  float a = 0.0f;
  #pragma unroll
  for (int k = 0; k < HDIM; ++k) a += qp[k] * (kp[k] + (float)rp[k]);
  sim[id] = a * INV_SQRT_HD;
}

__global__ __launch_bounds__(256) void scad_agg_kernel(
    const float* __restrict__ sim, const float* __restrict__ vn, const bf16* __restrict__ vr,
    const int* __restrict__ src, const int* __restrict__ cnt, const int* __restrict__ idx,
    float* __restrict__ agg, int n)
{
  long id = (long)blockIdx.x * 256 + threadIdx.x;
  if (id >= (long)n * NHD) return;
  int nd = (int)(id >> 3); int h = (int)(id & 7);
  float* out = agg + (long)nd * H + h * HDIM;
  int deg = cnt[nd]; if (deg > 8) deg = 8;
  if (deg == 0) {
    #pragma unroll
    for (int k = 0; k < HDIM; ++k) out[k] = 0.0f;
    return;
  }
  int ed[8];
  for (int j = 0; j < deg; ++j) ed[j] = idx[(long)nd * 8 + j];
  // sort edge ids -> deterministic FP reduction order
  for (int j = 1; j < deg; ++j) {
    int v = ed[j]; int p = j - 1;
    while (p >= 0 && ed[p] > v) { ed[p + 1] = ed[p]; --p; }
    ed[p + 1] = v;
  }
  float sv[8], m = -1e30f;
  for (int j = 0; j < deg; ++j) { sv[j] = sim[(long)ed[j] * 8 + h]; m = fmaxf(m, sv[j]); }
  float aw[8], den = 0.0f;
  for (int j = 0; j < deg; ++j) { aw[j] = expf(sv[j] - m); den += aw[j]; }
  float inv = 1.0f / (den + 1e-9f);
  float o[HDIM];
  #pragma unroll
  for (int k = 0; k < HDIM; ++k) o[k] = 0.0f;
  for (int j = 0; j < deg; ++j) {
    float wgt = aw[j] * inv;
    const float* vp = vn + (long)src[ed[j]] * H + h * HDIM;
    const bf16*  rp = vr + (long)ed[j] * H + h * HDIM;
    #pragma unroll
    for (int k = 0; k < HDIM; ++k) o[k] += wgt * (vp[k] + (float)rp[k]);
  }
  #pragma unroll
  for (int k = 0; k < HDIM; ++k) out[k] = o[k];
}

__global__ __launch_bounds__(256) void scad_gate_kernel(
    const float* __restrict__ G, const float* __restrict__ sn, float* __restrict__ agg, long n)
{
  long i = (long)blockIdx.x * 256 + threadIdx.x;
  if (i >= n) return;
  float g = 1.0f / (1.0f + expf(-G[i]));
  agg[i] += g * (sn[i] - agg[i]);
}

__global__ __launch_bounds__(256) void scad_cpt_kernel(
    const float* __restrict__ q, const float* __restrict__ kc, const float* __restrict__ vc,
    float* __restrict__ agg, int n)
{
  long id = (long)blockIdx.x * 256 + threadIdx.x;
  if (id >= (long)n * NHD) return;
  int nd = (int)(id >> 3); int h = (int)(id & 7);
  const float* qp = q + (long)nd * H + h * HDIM;
  float lg[16], m = -1e30f;
  for (int c = 0; c < 16; ++c) {
    const float* kp = kc + c * H + h * HDIM;
    float a = 0.0f;
    #pragma unroll
    for (int k = 0; k < HDIM; ++k) a += qp[k] * kp[k];
    lg[c] = a * INV_SQRT_HD;
    m = fmaxf(m, lg[c]);
  }
  float den = 0.0f;
  for (int c = 0; c < 16; ++c) { lg[c] = expf(lg[c] - m); den += lg[c]; }
  float inv = 1.0f / den;
  float o[HDIM];
  #pragma unroll
  for (int k = 0; k < HDIM; ++k) o[k] = 0.0f;
  for (int c = 0; c < 16; ++c) {
    float wgt = lg[c] * inv;
    const float* vp = vc + c * H + h * HDIM;
    #pragma unroll
    for (int k = 0; k < HDIM; ++k) o[k] += wgt * vp[k];
  }
  float* out = agg + (long)nd * H + h * HDIM;
  #pragma unroll
  for (int k = 0; k < HDIM; ++k) out[k] = o[k];
}

// ------------------------------------------------------------------
// Host side: param tree walker (JAX sorted-key flatten order) + pipeline
// ------------------------------------------------------------------
struct MLPp  { const float *b1,*b2,*be,*g,*w1,*w2; };
struct FOURp { const float *b1,*b2,*be1,*beo,*bo,*freqs,*g1,*go,*w1,*w2,*wo; };
struct ATTNp { const float *bf1,*bf2,*bg,*bk,*bkr,*bo,*bq,*bs,*bv,*bvr,
               *ln2_b,*ln2_g,*lns_b,*lns_g,
               *wf1,*wf2,*wg,*wk,*wkr,*wo,*wq,*ws,*wv,*wvr; };
struct CPTp  { const float *bf1,*bf2,*bk,*bo,*bq,*bv,*concepts,
               *ln2_b,*ln2_g,*ln_b,*ln_g,*wf1,*wf2,*wk,*wo,*wq,*wv; };

extern "C" void kernel_launch(void* const* d_in, const int* in_sizes, int n_in,
                              void* d_out, int out_size, void* d_ws, size_t ws_size,
                              hipStream_t stream) {
  (void)n_in; (void)out_size; (void)ws_size;
  const float* traj  = (const float*)d_in[0];
  const float* pos   = (const float*)d_in[1];
  const float* head  = (const float*)d_in[2];
  const float* props = (const float*)d_in[3];
  const float* P     = (const float*)d_in[4];
  const int* ati     = (const int*)d_in[5];
  const int* atype   = (const int*)d_in[6];
  const int* ustate  = (const int*)d_in[7];
  const int* owner   = (const int*)d_in[8];
  const int* et      = (const int*)d_in[9];
  const int* ea      = (const int*)d_in[10];
  const int Et = in_sizes[9] / 2;
  const int Ea = in_sizes[10] / 2;
  const int EM = Et > Ea ? Et : Ea;
  const int* srcT = et;        const int* dstT = et + Et;
  const int* srcA0 = ea;       const int* dstA0 = ea + Ea;

  // ---- params (JAX pytree: sorted dict keys, lists in order) ----
  size_t po = 0;
  auto take = [&](size_t n) { const float* r = P + po; po += n; return r; };
  auto takeMLP = [&](int i, int h, int o) {
    MLPp m; m.b1 = take(h); m.b2 = take(o); m.be = take(h); m.g = take(h);
    m.w1 = take((size_t)i * h); m.w2 = take((size_t)h * o); return m;
  };
  auto takeF = [&](int D) {
    FOURp f; f.b1 = take((size_t)D * H); f.b2 = take((size_t)D * H); f.be1 = take((size_t)D * H);
    f.beo = take(H); f.bo = take(H); f.freqs = take((size_t)D * FF); f.g1 = take((size_t)D * H);
    f.go = take(H); f.w1 = take((size_t)D * 129 * H); f.w2 = take((size_t)D * H * H);
    f.wo = take((size_t)H * H); return f;
  };
  auto takeA = [&]() {
    ATTNp a;
    a.bf1 = take(4 * H); a.bf2 = take(H); a.bg = take(H); a.bk = take(H); a.bkr = take(H);
    a.bo = take(H); a.bq = take(H); a.bs = take(H); a.bv = take(H); a.bvr = take(H);
    a.ln2_b = take(H); a.ln2_g = take(H); a.lns_b = take(H); a.lns_g = take(H);
    a.wf1 = take((size_t)H * 4 * H); a.wf2 = take((size_t)4 * H * H); a.wg = take((size_t)2 * H * H);
    a.wk = take((size_t)H * H); a.wkr = take((size_t)H * H); a.wo = take((size_t)H * H);
    a.wq = take((size_t)H * H); a.ws = take((size_t)H * H); a.wv = take((size_t)H * H);
    a.wvr = take((size_t)H * H); return a;
  };
  auto takeC = [&]() {
    CPTp c;
    c.bf1 = take(4 * H); c.bf2 = take(H); c.bk = take(H); c.bo = take(H); c.bq = take(H);
    c.bv = take(H); c.concepts = take(16 * H);
    c.ln2_b = take(H); c.ln2_g = take(H); c.ln_b = take(H); c.ln_g = take(H);
    c.wf1 = take((size_t)H * 4 * H); c.wf2 = take((size_t)4 * H * H);
    c.wk = take((size_t)H * H); c.wo = take((size_t)H * H); c.wq = take((size_t)H * H);
    c.wv = take((size_t)H * H); return c;
  };
  MLPp fusionp = takeMLP(2 * H, H, H);
  MLPp headp   = takeMLP(H, H, NTOKEN);
  ATTNp La2a[2]; CPTp Lcpt[2]; ATTNp Lt[2];
  for (int l = 0; l < 2; ++l) { La2a[l] = takeA(); Lcpt[l] = takeC(); Lt[l] = takeA(); }
  const float* owner_emb = take(3 * H);
  MLPp propsp = takeMLP(4, H, H);
  FOURp ra2ap = takeF(3);
  FOURp rtp   = takeF(4);
  const float* state_emb = take(4 * H);
  MLPp tokp = takeMLP(8, H, H);
  const float* type_emb = take(256 * H);
  FOURp xap = takeF(2);

  // ---- workspace ----
  char* W = (char*)d_ws;
  size_t wo = 0;
  auto alloc = [&](size_t bytes) { void* p = W + wo; wo += (bytes + 255) & ~(size_t)255; return p; };
  float* x  = (float*)alloc((size_t)NTOK * H * 4);
  float* xs = (float*)alloc((size_t)NTOK * H * 4);
  float* qb = (float*)alloc((size_t)NTOK * H * 4);   // also tok_emb
  float* kb = (float*)alloc((size_t)NTOK * H * 4);   // also x_a
  float* vb = (float*)alloc((size_t)NTOK * H * 4);   // also fusion pre-LN
  float* sb = (float*)alloc((size_t)NTOK * H * 4);
  float* ab = (float*)alloc((size_t)NTOK * H * 4);   // agg / fourier-xa accumulator
  float* Gb = (float*)alloc((size_t)NTOK * H * 4);
  float* bigS = (float*)alloc((size_t)EM * H * 4);   // fourier sum (f32) / FFN hidden [NTOK,1024]
  bf16*  bigH = (bf16*)alloc((size_t)EM * H * 2);    // fourier hidden (bf16) / kr-vr edge buffer
  bf16*  featd = (bf16*)alloc((size_t)EM * FLDA * 2);// fourier features; sim aliases this
  float* sim = (float*)featd;
  bf16* rT = (bf16*)alloc((size_t)Et * H * 2);
  bf16* rA = (bf16*)alloc((size_t)Ea * H * 2);
  int* cntT = (int*)alloc((size_t)NTOK * 4);
  int* idxT = (int*)alloc((size_t)NTOK * 8 * 4);
  int* cntA = (int*)alloc((size_t)NTOK * 4);
  int* idxA = (int*)alloc((size_t)NTOK * 8 * 4);
  int* srcA = (int*)alloc((size_t)Ea * 4);
  int* dstA = (int*)alloc((size_t)Ea * 4);
  float* rtfeat = (float*)alloc((size_t)Et * 4 * 4);
  float* rafeat = (float*)alloc((size_t)Ea * 3 * 4);
  float* tokTmp = (float*)alloc((size_t)NTOKEN * H * 4);
  float* tokAll = (float*)alloc((size_t)NTOKEN * H * 4);
  float* propsTmp = (float*)alloc((size_t)NA * H * 4);
  float* propsOut = (float*)alloc((size_t)NA * H * 4);
  float* feat = (float*)alloc((size_t)NTOK * 2 * 4);
  float* kcbuf = (float*)alloc((size_t)16 * H * 4);
  float* vcbuf = (float*)alloc((size_t)16 * H * 4);

  auto GEMM = [&](const void* A, int abf, long lda, const float* B, long ldb, const float* bias,
                  void* C, int cbf, long ldc, int M, int N, int K, int flags) {
    dim3 g((unsigned)((N + 63) / 64), (unsigned)((M + 127) / 128), 1);
    if (abf) {
      if (cbf) scad_gemm_kernel<1,1><<<g, 256, 0, stream>>>(A, lda, B, ldb, bias, C, ldc, M, N, K, flags);
      else     scad_gemm_kernel<1,0><<<g, 256, 0, stream>>>(A, lda, B, ldb, bias, C, ldc, M, N, K, flags);
    } else {
      if (cbf) scad_gemm_kernel<0,1><<<g, 256, 0, stream>>>(A, lda, B, ldb, bias, C, ldc, M, N, K, flags);
      else     scad_gemm_kernel<0,0><<<g, 256, 0, stream>>>(A, lda, B, ldb, bias, C, ldc, M, N, K, flags);
    }
  };
  auto LN = [&](const void* in, int ibf, void* out, int obf, const float* g, const float* b,
                int M, int relu) {
    dim3 gr((unsigned)((M + 7) / 8));
    if (ibf) {
      if (obf) scad_ln_kernel<1,1><<<gr, 256, 0, stream>>>(in, out, g, b, M, relu);
      else     scad_ln_kernel<1,0><<<gr, 256, 0, stream>>>(in, out, g, b, M, relu);
    } else {
      if (obf) scad_ln_kernel<0,1><<<gr, 256, 0, stream>>>(in, out, g, b, M, relu);
      else     scad_ln_kernel<0,0><<<gr, 256, 0, stream>>>(in, out, g, b, M, relu);
    }
  };
  auto EW = [&](long n) { return dim3((unsigned)((n + 255) / 256)); };

  // ---- tok MLP -> tokAll ----
  GEMM(traj, 0, 8, tokp.w1, H, tokp.b1, tokTmp, 0, H, NTOKEN, H, 8, 1);
  LN(tokTmp, 0, tokTmp, 0, tokp.g, tokp.be, NTOKEN, 1);
  GEMM(tokTmp, 0, H, tokp.w2, H, tokp.b2, tokAll, 0, H, NTOKEN, H, H, 1);
  // ---- props MLP -> propsOut ----
  GEMM(props, 0, 4, propsp.w1, H, propsp.b1, propsTmp, 0, H, NA, H, 4, 1);
  LN(propsTmp, 0, propsTmp, 0, propsp.g, propsp.be, NA, 1);
  GEMM(propsTmp, 0, H, propsp.w2, H, propsp.b2, propsOut, 0, H, NA, H, H, 1);
  // ---- motion features + token gather ----
  scad_feat_kernel<<<EW(NTOK), 256, 0, stream>>>(pos, head, feat);
  scad_gather_tok_kernel<<<EW((long)NTOK * H), 256, 0, stream>>>(tokAll, ati, qb);
  // ---- x_a = fourier(xa, feat, cat) -> kb ----
  for (int d = 0; d < 2; ++d) {
    scad_fourier_feat_kernel<<<EW((long)NTOK * FF), 256, 0, stream>>>(feat, 2, d, xap.freqs + d * FF, featd, NTOK);
    GEMM(featd, 1, FLDA, xap.w1 + (size_t)d * 129 * H, H, xap.b1 + d * H, bigH, 1, H, NTOK, H, 129, 1);
    LN(bigH, 1, bigH, 1, xap.g1 + d * H, xap.be1 + d * H, NTOK, 1);
    GEMM(bigH, 1, H, xap.w2 + (size_t)d * H * H, H, xap.b2 + d * H, ab, 0, H, NTOK, H, H, d == 0 ? 1 : 5);
  }
  scad_add_cat_kernel<<<EW((long)NTOK * H), 256, 0, stream>>>(ab, type_emb, state_emb, owner_emb,
                                                              propsOut, atype, ustate, owner);
  LN(ab, 0, Gb, 0, xap.go, xap.beo, NTOK, 1);
  GEMM(Gb, 0, H, xap.wo, H, xap.bo, kb, 0, H, NTOK, H, H, 1);
  // ---- fusion MLP -> x ----
  GEMM(qb, 0, H, fusionp.w1, H, fusionp.b1, vb, 0, H, NTOK, H, H, 1);
  GEMM(kb, 0, H, fusionp.w1 + (size_t)H * H, H, nullptr, vb, 0, H, NTOK, H, H, 4);
  LN(vb, 0, vb, 0, fusionp.g, fusionp.be, NTOK, 1);
  GEMM(vb, 0, H, fusionp.w2, H, fusionp.b2, x, 0, H, NTOK, H, H, 1);
  // ---- edge features ----
  scad_edge_feat_t_kernel<<<EW(Et), 256, 0, stream>>>(pos, head, srcT, dstT, Et, rtfeat);
  scad_edge_feat_a2a_kernel<<<EW(Ea), 256, 0, stream>>>(pos, head, srcA0, dstA0, Ea, rafeat, srcA, dstA);
  // ---- r_t = fourier(rt, rtfeat) -> rT (bf16) ----
  for (int d = 0; d < 4; ++d) {
    scad_fourier_feat_kernel<<<EW((long)Et * FF), 256, 0, stream>>>(rtfeat, 4, d, rtp.freqs + d * FF, featd, Et);
    GEMM(featd, 1, FLDA, rtp.w1 + (size_t)d * 129 * H, H, rtp.b1 + d * H, bigH, 1, H, Et, H, 129, 1);
    LN(bigH, 1, bigH, 1, rtp.g1 + d * H, rtp.be1 + d * H, Et, 1);
    GEMM(bigH, 1, H, rtp.w2 + (size_t)d * H * H, H, rtp.b2 + d * H, bigS, 0, H, Et, H, H, d == 0 ? 1 : 5);
  }
  LN(bigS, 0, bigH, 1, rtp.go, rtp.beo, Et, 1);
  GEMM(bigH, 1, H, rtp.wo, H, rtp.bo, rT, 1, H, Et, H, H, 1);
  // ---- r_a2a = fourier(ra2a, rafeat) -> rA (bf16) ----
  for (int d = 0; d < 3; ++d) {
    scad_fourier_feat_kernel<<<EW((long)Ea * FF), 256, 0, stream>>>(rafeat, 3, d, ra2ap.freqs + d * FF, featd, Ea);
    GEMM(featd, 1, FLDA, ra2ap.w1 + (size_t)d * 129 * H, H, ra2ap.b1 + d * H, bigH, 1, H, Ea, H, 129, 1);
    LN(bigH, 1, bigH, 1, ra2ap.g1 + d * H, ra2ap.be1 + d * H, Ea, 1);
    GEMM(bigH, 1, H, ra2ap.w2 + (size_t)d * H * H, H, ra2ap.b2 + d * H, bigS, 0, H, Ea, H, H, d == 0 ? 1 : 5);
  }
  LN(bigS, 0, bigH, 1, ra2ap.go, ra2ap.beo, Ea, 1);
  GEMM(bigH, 1, H, ra2ap.wo, H, ra2ap.bo, rA, 1, H, Ea, H, H, 1);
  // ---- CSR (degree <= 8 by construction) ----
  scad_zero_i32_kernel<<<EW(NTOK), 256, 0, stream>>>(cntT, NTOK);
  scad_csr_build_kernel<<<EW(Et), 256, 0, stream>>>(dstT, Et, cntT, idxT);
  scad_zero_i32_kernel<<<EW(NTOK), 256, 0, stream>>>(cntA, NTOK);
  scad_csr_build_kernel<<<EW(Ea), 256, 0, stream>>>(dstA, Ea, cntA, idxA);

  auto edgeAttn = [&](const ATTNp& A, bf16* r, const int* src, const int* dst,
                      const int* cnt, const int* idx, int E) {
    LN(x, 0, xs, 0, A.lns_g, A.lns_b, NTOK, 0);
    GEMM(xs, 0, H, A.wq, H, A.bq, qb, 0, H, NTOK, H, H, 1);
    GEMM(xs, 0, H, A.wk, H, A.bk, kb, 0, H, NTOK, H, H, 1);
    GEMM(xs, 0, H, A.wv, H, A.bv, vb, 0, H, NTOK, H, H, 1);
    GEMM(xs, 0, H, A.ws, H, A.bs, sb, 0, H, NTOK, H, H, 1);
    GEMM(r, 1, H, A.wkr, H, A.bkr, bigH, 1, H, E, H, H, 1);
    scad_sim_kernel<<<EW((long)E * NHD), 256, 0, stream>>>(qb, kb, bigH, src, dst, E, sim);
    GEMM(r, 1, H, A.wvr, H, A.bvr, bigH, 1, H, E, H, H, 1);
    scad_agg_kernel<<<EW((long)NTOK * NHD), 256, 0, stream>>>(sim, vb, bigH, src, cnt, idx, ab, NTOK);
    GEMM(ab, 0, H, A.wg, H, A.bg, Gb, 0, H, NTOK, H, H, 1);
    GEMM(xs, 0, H, A.wg + (size_t)H * H, H, nullptr, Gb, 0, H, NTOK, H, H, 4);
    scad_gate_kernel<<<EW((long)NTOK * H), 256, 0, stream>>>(Gb, sb, ab, (long)NTOK * H);
    GEMM(ab, 0, H, A.wo, H, A.bo, x, 0, H, NTOK, H, H, 5);
    LN(x, 0, xs, 0, A.ln2_g, A.ln2_b, NTOK, 0);
    GEMM(xs, 0, H, A.wf1, 4 * H, A.bf1, bigS, 0, 4 * H, NTOK, 4 * H, H, 3);
    GEMM(bigS, 0, 4 * H, A.wf2, H, A.bf2, x, 0, H, NTOK, H, 4 * H, 5);
  };
  auto cptAttn = [&](const CPTp& Cp) {
    LN(x, 0, xs, 0, Cp.ln_g, Cp.ln_b, NTOK, 0);
    GEMM(xs, 0, H, Cp.wq, H, Cp.bq, qb, 0, H, NTOK, H, H, 1);
    GEMM(Cp.concepts, 0, H, Cp.wk, H, Cp.bk, kcbuf, 0, H, 16, H, H, 1);
    GEMM(Cp.concepts, 0, H, Cp.wv, H, Cp.bv, vcbuf, 0, H, 16, H, H, 1);
    scad_cpt_kernel<<<EW((long)NTOK * NHD), 256, 0, stream>>>(qb, kcbuf, vcbuf, ab, NTOK);
    GEMM(ab, 0, H, Cp.wo, H, Cp.bo, x, 0, H, NTOK, H, H, 5);
    LN(x, 0, xs, 0, Cp.ln2_g, Cp.ln2_b, NTOK, 0);
    GEMM(xs, 0, H, Cp.wf1, 4 * H, Cp.bf1, bigS, 0, 4 * H, NTOK, 4 * H, H, 3);
    GEMM(bigS, 0, 4 * H, Cp.wf2, H, Cp.bf2, x, 0, H, NTOK, H, 4 * H, 5);
  };

  for (int l = 0; l < 2; ++l) {
    edgeAttn(Lt[l], rT, srcT, dstT, cntT, idxT, Et);
    edgeAttn(La2a[l], rA, srcA, dstA, cntA, idxA, Ea);
    cptAttn(Lcpt[l]);
  }

  // ---- head MLP -> d_out [NTOK, 1024] f32 ----
  GEMM(x, 0, H, headp.w1, H, headp.b1, xs, 0, H, NTOK, H, H, 1);
  LN(xs, 0, xs, 0, headp.g, headp.be, NTOK, 1);
  GEMM(xs, 0, H, headp.w2, NTOKEN, headp.b2, d_out, 0, NTOKEN, NTOK, NTOKEN, H, 1);
}
